// ASLVisionTransformer_63084479643854
// MI455X (gfx1250) — matrix operand order repaired and checked by reading, not compiled, not run on hardware
//
#include <hip/hip_runtime.h>
#include <hip/hip_bf16.h>
#include <cstdint>
#include <cstddef>

typedef __attribute__((ext_vector_type(16))) __bf16 bf16x16;
typedef __attribute__((ext_vector_type(8)))  float  f32x8;

#define S_LEN   8192
#define DIM_    1024
#define NH      16
#define NKV     4
#define HD      64
#define HIDDEN_ 4096
#define WIN     256

// ---------------- CDNA5 async global->LDS helpers ----------------
// GLOBAL_LOAD_ASYNC_TO_LDS_B128: per-lane copy of 16B memory -> LDS, tracked by
// ASYNCcnt; INST_OFFSET is applied to BOTH the global and the LDS address, so a
// single (lds_addr, global_addr) pair serves a whole contiguous per-thread chunk.

__device__ __forceinline__ void async_copy_32B(uint32_t lds_addr, const void* g) {
  asm volatile("global_load_async_to_lds_b128 %0, %1, off\n\t"
               "global_load_async_to_lds_b128 %0, %1, off offset:16"
               :: "v"(lds_addr), "v"(g) : "memory");
}

__device__ __forceinline__ void async_copy_64B(uint32_t lds_addr, const void* g) {
  asm volatile("global_load_async_to_lds_b128 %0, %1, off\n\t"
               "global_load_async_to_lds_b128 %0, %1, off offset:16\n\t"
               "global_load_async_to_lds_b128 %0, %1, off offset:32\n\t"
               "global_load_async_to_lds_b128 %0, %1, off offset:48"
               :: "v"(lds_addr), "v"(g) : "memory");
}

__device__ __forceinline__ void wait_async0() {
  asm volatile("s_wait_asynccnt 0x0" ::: "memory");
}

// generic LDS pointer -> wave-relative LDS byte address (low 32 bits of flat addr)
__device__ __forceinline__ uint32_t lds_addr_of(const void* p) {
  return (uint32_t)(uintptr_t)p;
}

// ---------------- conversion kernels ----------------

__global__ void k_f32_to_bf16(const float* __restrict__ in, __bf16* __restrict__ out, int n) {
  int i = blockIdx.x * 256 + threadIdx.x;
  if (i < n) out[i] = (__bf16)in[i];
}

// in: [K][N] f32 (row-major)  ->  out: [N][K] bf16 (row-major, i.e. B^T)
__global__ __launch_bounds__(256) void k_transpose_bf16(const float* __restrict__ in,
                                                        __bf16* __restrict__ out,
                                                        int K, int N) {
  __shared__ float tile[32][33];
  int k0 = blockIdx.y * 32, n0 = blockIdx.x * 32;
  int tx = threadIdx.x, ty = threadIdx.y;
  for (int i = ty; i < 32; i += 8)
    tile[i][tx] = in[(size_t)(k0 + i) * N + n0 + tx];
  __syncthreads();
  for (int i = ty; i < 32; i += 8)
    out[(size_t)(n0 + i) * K + k0 + tx] = (__bf16)tile[tx][i];
}

// ---------------- rmsnorm ----------------
__global__ __launch_bounds__(256) void k_rmsnorm(const float* __restrict__ x,
                                                 const float* __restrict__ w,
                                                 __bf16* __restrict__ outB,
                                                 float* __restrict__ outF) {
  __shared__ float red[256];
  size_t row = blockIdx.x;
  const float* xr = x + row * DIM_;
  float v[4];
  float s = 0.f;
#pragma unroll
  for (int j = 0; j < 4; ++j) {
    v[j] = xr[threadIdx.x + j * 256];
    s += v[j] * v[j];
  }
  red[threadIdx.x] = s;
  __syncthreads();
  for (int off = 128; off > 0; off >>= 1) {
    if (threadIdx.x < off) red[threadIdx.x] += red[threadIdx.x + off];
    __syncthreads();
  }
  float rs = rsqrtf(red[0] * (1.0f / DIM_) + 1e-6f);
#pragma unroll
  for (int j = 0; j < 4; ++j) {
    int c = threadIdx.x + j * 256;
    float y = v[j] * rs * w[c];
    if (outB) outB[row * DIM_ + c] = (__bf16)y;
    if (outF) outF[row * DIM_ + c] = y;
  }
}

// ---------------- rope (in place) ----------------
__global__ void k_rope(float* __restrict__ x, int nheads, int total) {
  int tid = blockIdx.x * 256 + threadIdx.x;
  if (tid >= total) return;
  int i = tid & 31;
  int h = (tid >> 5) % nheads;
  int s = tid / (32 * nheads);
  float inv = __powf(10000.0f, -(float)i * (1.0f / 32.0f));
  float ang = (float)s * inv;
  float sn, cs;
  __sincosf(ang, &sn, &cs);
  float* p = x + ((size_t)s * nheads + h) * HD + 2 * i;
  float x0 = p[0], x1 = p[1];
  p[0] = x0 * cs - x1 * sn;
  p[1] = x0 * sn + x1 * cs;
}

// ---------------- sliding-window attention ----------------
// Q: [S][16][64] f32 (rope'd), K/V: [S][4][64] f32. One thread = one query row.
// K/V tiles staged via async global->LDS. Output bf16 [S][1024] for the wo GEMM.
__global__ __launch_bounds__(256) void k_attn(const float* __restrict__ Q,
                                              const float* __restrict__ K,
                                              const float* __restrict__ V,
                                              __bf16* __restrict__ O) {
  __shared__ float sk[64][64];
  __shared__ float sv[64][64];
  int c = blockIdx.x;           // chunk 0..31
  int h = blockIdx.y;           // head 0..15
  int kvh = h >> 2;             // GQA: 4 query heads per kv head
  int qi = threadIdx.x;         // query index in chunk

  const float* qp = Q + ((size_t)(c * WIN + qi) * NH + h) * HD;
  float qr[HD];
#pragma unroll
  for (int d = 0; d < HD; ++d) qr[d] = qp[d];

  float m = -3.0e38f, l = 0.f;
  float acc[HD];
#pragma unroll
  for (int d = 0; d < HD; ++d) acc[d] = 0.f;

  int jr = threadIdx.x >> 2;             // key row this thread stages
  int dd = (threadIdx.x & 3) * 16;       // 16-float chunk (64B)
  uint32_t sko = lds_addr_of(&sk[jr][dd]);
  uint32_t svo = lds_addr_of(&sv[jr][dd]);

  for (int jt = 0; jt < 8; ++jt) {
    int kcat0 = jt * 64;                 // position in [prev(256) | cur(256)]
    int g0 = c * WIN - WIN + kcat0;      // global sequence position of tile start
    __syncthreads();                     // previous tile fully consumed
    if (g0 < 0) continue;                // chunk 0: previous chunk is masked out
    async_copy_64B(sko, K + ((size_t)(g0 + jr) * NKV + kvh) * HD + dd);
    async_copy_64B(svo, V + ((size_t)(g0 + jr) * NKV + kvh) * HD + dd);
    wait_async0();
    __syncthreads();
    for (int j = 0; j < 64; ++j) {
      int kcat = kcat0 + j;
      // mask: kcat in [qi+1, qi+256]  (causal window of 256 incl. self at qi+256)
      if (kcat < qi + 1 || kcat > qi + WIN) continue;
      float sdot = 0.f;
#pragma unroll
      for (int d = 0; d < HD; ++d) sdot += qr[d] * sk[j][d];
      sdot *= 0.125f;  // 64^-0.5
      float nm = fmaxf(m, sdot);
      float corr = __expf(m - nm);
      float p = __expf(sdot - nm);
      l = l * corr + p;
#pragma unroll
      for (int d = 0; d < HD; ++d) acc[d] = acc[d] * corr + p * sv[j][d];
      m = nm;
    }
  }
  float invl = 1.0f / l;  // self is always attended, l > 0
  __bf16* op = O + ((size_t)(c * WIN + qi) * NH + h) * HD;
#pragma unroll
  for (int d = 0; d < HD; ++d) op[d] = (__bf16)(acc[d] * invl);
}

// ---------------- WMMA bf16 GEMM (double-buffered, async tile copies) ----------------
// out[M][N] = A[M][K](bf16,row-major) x Bt[N][K](bf16, B transposed) [+bias] [+addsrc]
// mode 0: outF = acc (+bias[col]) (+addsrc[idx])      (f32)
// mode 1: outB = acc                                  (bf16)
// mode 2: outB = silu(other[idx]) * acc               (bf16; fused SwiGLU gate)
#define BM  128
#define BN  128
#define BK  32
#define LDA 40   // 32 + 8 bf16 pad; row stride 80B keeps 16B alignment

union BF16Frag { bf16x16 v; uint4 u[2]; };

__global__ __launch_bounds__(256) void k_gemm(const __bf16* __restrict__ A,
                                              const __bf16* __restrict__ Bt,
                                              const float* __restrict__ bias,
                                              const float* __restrict__ addsrc,
                                              const float* __restrict__ other,
                                              float* __restrict__ outF,
                                              __bf16* __restrict__ outB,
                                              int M, int N, int K, int mode) {
  __shared__ __bf16 sA[2 * BM * LDA];
  __shared__ __bf16 sB[2 * BN * LDA];
  const uint32_t BUFBYTES = BM * LDA * sizeof(__bf16);  // 10240B per buffer

  int t = threadIdx.x;
  int bm = blockIdx.y * BM;
  int bn = blockIdx.x * BN;
  int lane = t & 31, wave = t >> 5;
  int wm = wave & 1, wn = wave >> 1;        // 2 x 4 wave grid -> 64x32 per wave
  int lrow = lane & 15, lhi = lane >> 4;
  int arow = t >> 1, ahalf = t & 1;         // 2 threads per 32-elem row

  const __bf16* gA = A + (size_t)(bm + arow) * K + ahalf * 16;
  const __bf16* gB = Bt + (size_t)(bn + arow) * K + ahalf * 16;
  uint32_t stA = lds_addr_of(sA + arow * LDA + ahalf * 16);
  uint32_t stB = lds_addr_of(sB + arow * LDA + ahalf * 16);

  f32x8 acc[4][2];
#pragma unroll
  for (int mi = 0; mi < 4; ++mi)
#pragma unroll
    for (int ni = 0; ni < 2; ++ni)
#pragma unroll
      for (int e = 0; e < 8; ++e) acc[mi][ni][e] = 0.f;

  // prologue: stage K-tile 0 into buffer 0 (async, no VGPR staging)
  async_copy_32B(stA, gA);
  async_copy_32B(stB, gB);

  int nk = K / BK;
  for (int kt = 0; kt < nk; ++kt) {
    int cur = kt & 1;
    wait_async0();        // this wave's async writes into buf[cur] are in LDS
    __syncthreads();      // all waves' writes visible; all prior reads done
    if (kt + 1 < nk) {    // prefetch next K-tile into the other buffer
      int nxt = cur ^ 1;
      async_copy_32B(stA + nxt * BUFBYTES, gA + (size_t)(kt + 1) * BK);
      async_copy_32B(stB + nxt * BUFBYTES, gB + (size_t)(kt + 1) * BK);
    }
    const __bf16* cA = sA + cur * (BM * LDA);
    const __bf16* cB = sB + cur * (BM * LDA);

    BF16Frag af[4], bf[2];
#pragma unroll
    for (int mi = 0; mi < 4; ++mi) {
      // A 16x32 layout: lanes 0-15 hold K 0-7 & 16-23; lanes 16-31 hold K 8-15 & 24-31
      const __bf16* base = cA + (wm * 64 + mi * 16 + lrow) * LDA;
      af[mi].u[0] = *(const uint4*)(base + lhi * 8);
      af[mi].u[1] = *(const uint4*)(base + lhi * 8 + 16);
    }
#pragma unroll
    for (int ni = 0; ni < 2; ++ni) {
      // B 32x16 layout: lanes 0-15 hold K 0-7 & 16-23; lanes 16-31 hold K 8-15 & 24-31 (col = lane%16)
      const __bf16* base = cB + (wn * 32 + ni * 16 + lrow) * LDA;
      bf[ni].u[0] = *(const uint4*)(base + lhi * 8);
      bf[ni].u[1] = *(const uint4*)(base + lhi * 8 + 16);
    }
#pragma unroll
    for (int mi = 0; mi < 4; ++mi)
#pragma unroll
      for (int ni = 0; ni < 2; ++ni)
        acc[mi][ni] = __builtin_amdgcn_wmma_f32_16x16x32_bf16(
            false, af[mi].v, false, bf[ni].v, (short)0, acc[mi][ni], false, false);
    // no trailing barrier: next iteration's wait+barrier precedes buffer reuse
  }

  // epilogue: C/D layout: VGPR r -> M = r + 8*lhi, N = lane%16
#pragma unroll
  for (int mi = 0; mi < 4; ++mi) {
#pragma unroll
    for (int ni = 0; ni < 2; ++ni) {
      int gcol = bn + wn * 32 + ni * 16 + lrow;
#pragma unroll
      for (int r = 0; r < 8; ++r) {
        int grow = bm + wm * 64 + mi * 16 + lhi * 8 + r;
        size_t idx = (size_t)grow * N + gcol;
        float v = acc[mi][ni][r];
        if (mode == 0) {
          if (bias)   v += bias[gcol];
          if (addsrc) v += addsrc[idx];
          outF[idx] = v;
        } else if (mode == 1) {
          outB[idx] = (__bf16)v;
        } else {
          float g = other[idx];
          float sg = g * (1.0f / (1.0f + __expf(-g)));
          outB[idx] = (__bf16)(sg * v);
        }
      }
    }
  }
}

// ---------------- pooling + classifier head ----------------

__global__ void k_pool(const float* __restrict__ hn, float* __restrict__ pooled) {
  int d = blockIdx.x * 256 + threadIdx.x;
  float s = 0.f;
  for (int r = 0; r < S_LEN; ++r) s += hn[(size_t)r * DIM_ + d];
  pooled[d] = s * (1.0f / S_LEN);
}

__global__ __launch_bounds__(256) void k_head(const float* __restrict__ pooled,
                                              const float* __restrict__ W,
                                              const float* __restrict__ b,
                                              float* __restrict__ out, int NC) {
  __shared__ float sp[DIM_];
  for (int i = threadIdx.x; i < DIM_; i += 256) sp[i] = pooled[i];
  __syncthreads();
  int n = blockIdx.x * 256 + threadIdx.x;
  if (n < NC) {
    float s = b[n];
    for (int k = 0; k < DIM_; ++k) s += sp[k] * W[(size_t)k * NC + n];
    out[n] = s;
  }
}

// ---------------- host orchestration ----------------

extern "C" void kernel_launch(void* const* d_in, const int* in_sizes, int n_in,
                              void* d_out, int out_size, void* d_ws, size_t ws_size,
                              hipStream_t stream) {
  (void)in_sizes; (void)n_in; (void)out_size; (void)ws_size;

  const float* x     = (const float*)d_in[0];
  // d_in[1] = seqlens (unused by reference)
  const float* emb_w = (const float*)d_in[2];
  const float* emb_b = (const float*)d_in[3];
  const float *wq[4], *wk[4], *wv[4], *wo[4], *w1[4], *w2[4], *w3[4], *anorm[4], *fnorm[4];
  for (int i = 0; i < 4; ++i) {
    int b = 4 + 9 * i;
    wq[i] = (const float*)d_in[b + 0];
    wk[i] = (const float*)d_in[b + 1];
    wv[i] = (const float*)d_in[b + 2];
    wo[i] = (const float*)d_in[b + 3];
    w1[i] = (const float*)d_in[b + 4];
    w2[i] = (const float*)d_in[b + 5];
    w3[i] = (const float*)d_in[b + 6];
    anorm[i] = (const float*)d_in[b + 7];
    fnorm[i] = (const float*)d_in[b + 8];
  }
  const float* norm_w = (const float*)d_in[40];
  const float* out_w  = (const float*)d_in[41];
  const float* out_b  = (const float*)d_in[42];

  // workspace carving (256B aligned)
  size_t off = 0;
  auto carve = [&](size_t bytes) -> void* {
    void* p = (char*)d_ws + off;
    off += (bytes + 255) & ~(size_t)255;
    return p;
  };
  __bf16* embT = (__bf16*)carve((size_t)256 * 1024 * 2);
  __bf16 *wqT[4], *wkT[4], *wvT[4], *woT[4], *w1T[4], *w2T[4], *w3T[4];
  for (int i = 0; i < 4; ++i) {
    wqT[i] = (__bf16*)carve((size_t)1024 * 1024 * 2);
    wkT[i] = (__bf16*)carve((size_t)1024 * 256 * 2);
    wvT[i] = (__bf16*)carve((size_t)1024 * 256 * 2);
    woT[i] = (__bf16*)carve((size_t)1024 * 1024 * 2);
    w1T[i] = (__bf16*)carve((size_t)1024 * 4096 * 2);
    w2T[i] = (__bf16*)carve((size_t)4096 * 1024 * 2);
    w3T[i] = (__bf16*)carve((size_t)1024 * 4096 * 2);
  }
  float*  h   = (float*)carve((size_t)S_LEN * DIM_ * 4);
  __bf16* xb  = (__bf16*)carve((size_t)S_LEN * DIM_ * 2);
  float*  Qb  = (float*)carve((size_t)S_LEN * DIM_ * 4);
  float*  Kb  = (float*)carve((size_t)S_LEN * 256 * 4);
  float*  Vb  = (float*)carve((size_t)S_LEN * 256 * 4);
  float*  G1  = (float*)carve((size_t)S_LEN * HIDDEN_ * 4);
  __bf16* GB  = (__bf16*)carve((size_t)S_LEN * HIDDEN_ * 2);
  float*  pooled = (float*)carve((size_t)DIM_ * 4);

  dim3 tb(32, 8);
  auto transT = [&](const float* src, __bf16* dst, int K, int N) {
    k_transpose_bf16<<<dim3(N / 32, K / 32), tb, 0, stream>>>(src, dst, K, N);
  };
  // weight convert + transpose (once per launch; deterministic)
  transT(emb_w, embT, 256, 1024);
  for (int i = 0; i < 4; ++i) {
    transT(wq[i], wqT[i], 1024, 1024);
    transT(wk[i], wkT[i], 1024, 256);
    transT(wv[i], wvT[i], 1024, 256);
    transT(wo[i], woT[i], 1024, 1024);
    transT(w1[i], w1T[i], 1024, 4096);
    transT(w2[i], w2T[i], 4096, 1024);
    transT(w3[i], w3T[i], 1024, 4096);
  }

  auto gemm = [&](const __bf16* A, const __bf16* Bt, const float* bias, const float* add,
                  const float* other, float* oF, __bf16* oB, int M, int N, int K, int mode) {
    k_gemm<<<dim3(N / BN, M / BM), 256, 0, stream>>>(A, Bt, bias, add, other, oF, oB, M, N, K, mode);
  };

  // embedding: h = bf16(x) @ emb_w + emb_b    (x bf16 staged in GB, free until FFN)
  {
    int nx = S_LEN * 256;
    k_f32_to_bf16<<<(nx + 255) / 256, 256, 0, stream>>>(x, GB, nx);
    gemm(GB, embT, emb_b, nullptr, nullptr, h, nullptr, S_LEN, 1024, 256, 0);
  }

  for (int i = 0; i < 4; ++i) {
    // attention block
    k_rmsnorm<<<S_LEN, 256, 0, stream>>>(h, anorm[i], xb, nullptr);
    gemm(xb, wqT[i], nullptr, nullptr, nullptr, Qb, nullptr, S_LEN, 1024, 1024, 0);
    gemm(xb, wkT[i], nullptr, nullptr, nullptr, Kb, nullptr, S_LEN, 256, 1024, 0);
    gemm(xb, wvT[i], nullptr, nullptr, nullptr, Vb, nullptr, S_LEN, 256, 1024, 0);
    {
      int tq = S_LEN * NH * 32, tk = S_LEN * NKV * 32;
      k_rope<<<(tq + 255) / 256, 256, 0, stream>>>(Qb, NH, tq);
      k_rope<<<(tk + 255) / 256, 256, 0, stream>>>(Kb, NKV, tk);
    }
    k_attn<<<dim3(S_LEN / WIN, NH), 256, 0, stream>>>(Qb, Kb, Vb, xb);  // xb <- bf16 attn out
    gemm(xb, woT[i], nullptr, h, nullptr, h, nullptr, S_LEN, 1024, 1024, 0);  // h += o @ wo

    // FFN block (SwiGLU), gate fused into second GEMM epilogue
    k_rmsnorm<<<S_LEN, 256, 0, stream>>>(h, fnorm[i], xb, nullptr);
    gemm(xb, w1T[i], nullptr, nullptr, nullptr, G1, nullptr, S_LEN, 4096, 1024, 0);   // G1 = f@w1
    gemm(xb, w3T[i], nullptr, nullptr, G1, nullptr, GB, S_LEN, 4096, 1024, 2);        // GB = silu(G1)*(f@w3)
    gemm(GB, w2T[i], nullptr, h, nullptr, h, nullptr, S_LEN, 1024, 4096, 0);          // h += GB @ w2
  }

  // final norm -> mean pool -> classifier
  k_rmsnorm<<<S_LEN, 256, 0, stream>>>(h, norm_w, nullptr, Qb);  // Qb reused as f32 normed h
  k_pool<<<DIM_ / 256, 256, 0, stream>>>(Qb, pooled);
  k_head<<<(2000 + 255) / 256, 256, 0, stream>>>(pooled, out_w, out_b, (float*)d_out, 2000);
}
